// SingleScaleGrid_65541200937490
// MI455X (gfx1250) — matrix operand order repaired and checked by reading, not compiled, not run on hardware
//
#include <hip/hip_runtime.h>
#include <hip/hip_bf16.h>

typedef __attribute__((ext_vector_type(16))) _Float16 v16h;
typedef __attribute__((ext_vector_type(8)))  _Float16 v8h;
typedef __attribute__((ext_vector_type(4)))  _Float16 v4h;
typedef __attribute__((ext_vector_type(8)))  float    v8f;

#define WMMA16(A,B,C) __builtin_amdgcn_wmma_f32_16x16x32_f16(false,(A),false,(B),(short)0,(C),false,false)

// Repack two f32 C tiles (channels [0..15] and [16..31] of a 32-chunk) into one
// f16 B fragment (B layout: element e <-> K = kh*16+e). Needs a half-wave swap.
__device__ __forceinline__ v16h pack_pair_shfl(const v8f cx, const v8f cy, int lane)
{
  const int base = lane & 15;
  const bool odd = lane >= 16;
  v16h b;
#pragma unroll
  for (int e = 0; e < 8; ++e) {
    float x0 = __shfl(cx[e], base);
    float y0 = __shfl(cy[e], base);
    float x1 = __shfl(cx[e], base + 16);
    float y1 = __shfl(cy[e], base + 16);
    b[e]     = (_Float16)(odd ? y0 : x0);
    b[e + 8] = (_Float16)(odd ? y1 : x1);
  }
  return b;
}

// ---------------------------------------------------------------------------
// K1: grid regularizer (3 conv1x1 GEMMs via WMMA) -> scatter x3 into fine grid
// feat[fineVoxel][16] f16, fine = (2i+a,2j+b,2k+d), ch l, c3 = 8l+4a+2b+d
// BN scale folded into A-fragment weights; epilogue = bias-init + PReLU only.
// One tile per wave, straight-line body (avoids LICM register blowup).
// ---------------------------------------------------------------------------
__global__ __launch_bounds__(256) void k_gridreg(
    const float* __restrict__ grid,
    const float* __restrict__ c1w, const float* __restrict__ c1b,
    const float* __restrict__ g1,  const float* __restrict__ bb1, const float* __restrict__ a1,
    const float* __restrict__ c2w, const float* __restrict__ c2b,
    const float* __restrict__ g2,  const float* __restrict__ bb2, const float* __restrict__ a2,
    const float* __restrict__ c3w, const float* __restrict__ c3b,
    _Float16* __restrict__ feat)
{
  __shared__ __align__(32) _Float16 wf[56][32][16];  // A fragments (f16, BN-folded)
  __shared__ float prm[384];                         // bias1'(64) pa1(64) bias2'(64) pa2(64) c3b(128)
  const float BNS = 0.99999500003749978f;
  const int tid = threadIdx.x;
  for (int i = tid; i < 64; i += 256) {
    prm[i]       = c1b[i] * (BNS * g1[i]) + bb1[i];  // folded bias layer1
    prm[64 + i]  = a1[i];
    prm[128 + i] = c2b[i] * (BNS * g2[i]) + bb2[i];  // folded bias layer2
    prm[192 + i] = a2[i];
  }
  for (int i = tid; i < 128; i += 256) prm[256 + i] = c3b[i];
  for (int slot = tid; slot < 56 * 32; slot += 256) {
    const int f = slot >> 5, ln = slot & 31;
    const int m16 = ln & 15, khw = ln >> 4;
    int which, mt, ks;
    if (f < 4)       { which = 0; mt = f;            ks = 0;            }
    else if (f < 16) { which = 1; mt = (f - 4) / 3;  ks = (f - 4) % 3;  }
    else             { which = 2; mt = (f - 16) / 5; ks = (f - 16) % 5; }
#pragma unroll
    for (int e = 0; e < 16; ++e) {
      const int kl = khw * 16 + e;
      const int k  = ks * 32 + kl;
      const int m  = mt * 16 + m16;
      float vv = 0.0f;
      if (which == 0)      { if (k < 16)  vv = c1w[m * 16  + k] * (BNS * g1[m]); }
      else if (which == 1) { if (k < 80)  vv = c2w[m * 80  + k] * (BNS * g2[m]); }
      else                 { if (k < 144) vv = c3w[m * 144 + k]; }
      wf[f][ln][e] = (_Float16)vv;
    }
  }
  __syncthreads();

  const int lane = tid & 31, wave = tid >> 5;
  const int n  = lane & 15, kh = lane >> 4;
  const int tile = blockIdx.x * 8 + wave;            // 2048*8 = 16384 tiles
  const int vb = tile * 16;
  const int v  = vb + n;

  // grid B fragment: B[k][n], k<16 real (kh==0 half), rest zero
  v16h bG;
#pragma unroll
  for (int e = 0; e < 16; ++e)
    bG[e] = (kh == 0) ? (_Float16)grid[e * 262144 + v] : (_Float16)0.0f;

  v8f c[4];
  // ---- GEMM1: x1 = prelu(W1'*G + bias1') ----
#pragma unroll
  for (int mt = 0; mt < 4; ++mt)
#pragma unroll
    for (int r = 0; r < 8; ++r) c[mt][r] = prm[mt * 16 + kh * 8 + r];
#pragma unroll
  for (int mt = 0; mt < 4; ++mt) {
    v16h a = *(const v16h*)&wf[mt][lane][0];
    c[mt] = WMMA16(a, bG, c[mt]);
  }
#pragma unroll
  for (int mt = 0; mt < 4; ++mt)
#pragma unroll
    for (int r = 0; r < 8; ++r) {
      const float pa = prm[64 + mt * 16 + kh * 8 + r];
      const float y = c[mt][r];
      c[mt][r] = fmaxf(y, 0.0f) + pa * fminf(y, 0.0f);   // branch-free PReLU
    }
  const v16h b0 = pack_pair_shfl(c[0], c[1], lane);
  const v16h b1 = pack_pair_shfl(c[2], c[3], lane);

  // ---- GEMM2: x2 over [x1; grid] ----
#pragma unroll
  for (int mt = 0; mt < 4; ++mt)
#pragma unroll
    for (int r = 0; r < 8; ++r) c[mt][r] = prm[128 + mt * 16 + kh * 8 + r];
#pragma unroll
  for (int mt = 0; mt < 4; ++mt)
#pragma unroll
    for (int ks = 0; ks < 3; ++ks) {
      v16h a = *(const v16h*)&wf[4 + mt * 3 + ks][lane][0];
      c[mt] = WMMA16(a, (ks == 0) ? b0 : (ks == 1) ? b1 : bG, c[mt]);
    }
#pragma unroll
  for (int mt = 0; mt < 4; ++mt)
#pragma unroll
    for (int r = 0; r < 8; ++r) {
      const float pa = prm[192 + mt * 16 + kh * 8 + r];
      const float y = c[mt][r];
      c[mt][r] = fmaxf(y, 0.0f) + pa * fminf(y, 0.0f);
    }
  const v16h b2 = pack_pair_shfl(c[0], c[1], lane);
  const v16h b3 = pack_pair_shfl(c[2], c[3], lane);

  // ---- GEMM3: x3 over [x1; x2; grid], scatter to fine grid as f16 ----
  const int vi = v >> 12, vj = (v >> 6) & 63, vk = v & 63;
#pragma unroll
  for (int mt = 0; mt < 8; ++mt) {
    v8f d;
#pragma unroll
    for (int r = 0; r < 8; ++r) d[r] = prm[256 + mt * 16 + kh * 8 + r];
#pragma unroll
    for (int ks = 0; ks < 5; ++ks) {
      v16h a = *(const v16h*)&wf[16 + mt * 5 + ks][lane][0];
      d = WMMA16(a, (ks==0)?b0:(ks==1)?b1:(ks==2)?b2:(ks==3)?b3:bG, d);
    }
    const int ch = 2 * mt + kh;                    // latent channel l
#pragma unroll
    for (int r = 0; r < 8; ++r) {
      const int aa = (r >> 2) & 1, bb = (r >> 1) & 1, dd = r & 1;
      const long fine = ((long)(2*vi+aa) * 128 + (2*vj+bb)) * 128 + (2*vk+dd);
      feat[fine * 16 + ch] = (_Float16)d[r];
    }
  }
}

// ---------------------------------------------------------------------------
// K2: separable 5x5x5 Sobel on density (feat ch0), normals = -normalize(sob)
//     Halo staged via async global->LDS copies (ASYNCcnt path).
// ---------------------------------------------------------------------------
__global__ __launch_bounds__(512) void k_sobel(const _Float16* __restrict__ feat,
                                               _Float16* __restrict__ nrm)
{
  __shared__ uint32_t t32[1728];                       // 12x12x12, f16 density in low half
  const int bx = blockIdx.x & 15, by = (blockIdx.x >> 4) & 15, bz = blockIdx.x >> 8;
  const int ox = bx*8 - 2, oy = by*8 - 2, oz = bz*8 - 2;

  // zero-fill (provides the conv zero padding), publish before async overwrite
  for (int s = threadIdx.x; s < 1728; s += 512) t32[s] = 0u;
  __syncthreads();

  // async copy of in-bounds halo cells: LDS[t32+s*4] <- feat[cell][ch0..1] (b32)
  for (int s = threadIdx.x; s < 1728; s += 512) {
    const int z = s % 12, y = (s / 12) % 12, x = s / 144;
    const int X = ox + x, Y = oy + y, Z = oz + z;
    if ((unsigned)X < 128u && (unsigned)Y < 128u && (unsigned)Z < 128u) {
      const uint32_t ldsa = (uint32_t)(uintptr_t)(&t32[0]) + (uint32_t)(s * 4);
      const uint32_t goff = (uint32_t)((((X * 128 + Y) * 128 + Z)) * 32);  // bytes
      asm volatile("global_load_async_to_lds_b32 %0, %1, %2"
                   :: "v"(ldsa), "v"(goff), "s"(feat) : "memory");
    }
  }
  asm volatile("s_wait_asynccnt 0x0" ::: "memory");
  __syncthreads();

  // convert f16 -> f32 in place
  for (int s = threadIdx.x; s < 1728; s += 512) {
    const unsigned short hraw = (unsigned short)(t32[s] & 0xFFFFu);
    const float fval = (float)__builtin_bit_cast(_Float16, hraw);
    t32[s] = __builtin_bit_cast(uint32_t, fval);
  }
  __syncthreads();

  const int lz = threadIdx.x & 7, ly = (threadIdx.x >> 3) & 7, lx = threadIdx.x >> 6;
  const float h[5]  = {1.f, 4.f, 6.f, 4.f, 1.f};
  const float hd[5] = {-1.f, -2.f, 0.f, 2.f, 1.f};
  float gx = 0.f, gy = 0.f, gz = 0.f;
#pragma unroll
  for (int i = 0; i < 5; ++i) {
    float Ai = 0.f, Bi = 0.f, Ci = 0.f;
#pragma unroll
    for (int j = 0; j < 5; ++j) {
      float sh = 0.f, sd = 0.f;
#pragma unroll
      for (int k = 0; k < 5; ++k) {
        const float d = __builtin_bit_cast(float, t32[((lx+i)*12 + (ly+j))*12 + (lz+k)]);
        sh += h[k] * d; sd += hd[k] * d;
      }
      Ai += h[j] * sh; Bi += hd[j] * sh; Ci += h[j] * sd;
    }
    gx += hd[i] * Ai; gy += h[i] * Bi; gz += h[i] * Ci;
  }
  const float inv72 = 1.0f / 72.0f;
  gx *= inv72; gy *= inv72; gz *= inv72;
  const float nn = sqrtf(gx*gx + gy*gy + gz*gz);
  const float inv = -1.0f / fmaxf(nn, 1e-12f);
  const int X = bx*8 + lx, Y = by*8 + ly, Z = bz*8 + lz;
  const long o = (((long)X * 128 + Y) * 128 + Z) * 4;
  nrm[o + 0] = (_Float16)(gx * inv);
  nrm[o + 1] = (_Float16)(gy * inv);
  nrm[o + 2] = (_Float16)(gz * inv);
  nrm[o + 3] = (_Float16)0.0f;
}

// ---------------------------------------------------------------------------
// K3: ray march. 1 ray / block, 4 segment-waves of 128 samples, 32 samples /
// iter, MLP via WMMA (f16), segmented transmittance composition in LDS.
// ---------------------------------------------------------------------------
__global__ __launch_bounds__(128) void k_render(
    const float* __restrict__ ro, const float* __restrict__ rd, const float* __restrict__ vd,
    const _Float16* __restrict__ feat, const _Float16* __restrict__ nrm,
    const float* __restrict__ w0, const float* __restrict__ b0,
    const float* __restrict__ w1, const float* __restrict__ b1,
    const float* __restrict__ w2, const float* __restrict__ b2,
    const float* __restrict__ w3, const float* __restrict__ b3,
    float* __restrict__ out)
{
  __shared__ __align__(32) _Float16 wf[26][32][16];
  __shared__ __align__(32) _Float16 hbuf[4][32][64];
  __shared__ float bias[208];
  __shared__ float comb[4][12];
  const int tid = threadIdx.x;
  for (int i = tid; i < 64; i += 128) { bias[i] = b0[i]; bias[64+i] = b1[i]; bias[128+i] = b2[i]; }
  if (tid < 16) bias[192 + tid] = (tid < 3) ? b3[tid] : 0.0f;
  for (int slot = tid; slot < 26 * 32; slot += 128) {
    const int f = slot >> 5, ln = slot & 31;
    const int m16 = ln & 15, khw = ln >> 4;
    int layer, mt, ks;
    if (f < 8)       { layer = 0; mt = f >> 1;        ks = f & 1;  }
    else if (f < 16) { layer = 1; mt = (f - 8) >> 1;  ks = f & 1;  }
    else if (f < 24) { layer = 2; mt = (f - 16) >> 1; ks = f & 1;  }
    else             { layer = 3; mt = 0;             ks = f - 24; }
    const float* W = (layer == 0) ? w0 : (layer == 1) ? w1 : (layer == 2) ? w2 : w3;
    const int Kin = (layer == 0) ? 54 : 64;
    const int Mo  = (layer == 3) ? 3 : 64;
    const int m = mt * 16 + m16;
#pragma unroll
    for (int e = 0; e < 16; ++e) {
      const int kl = khw * 16 + e;
      const int k = ks * 32 + kl;
      float vv = 0.0f;
      if (k < Kin && m < Mo) vv = W[k * Mo + m];
      wf[f][ln][e] = (_Float16)vv;
    }
  }
  __syncthreads();

  const int lane = tid & 31, wave = tid >> 5;   // wave = sample segment 0..3
  const int n16 = lane & 15, kh = lane >> 4, kh8 = kh * 8;
  const int ray = blockIdx.x;

  const float ox = ro[ray*3+0], oy = ro[ray*3+1], oz = ro[ray*3+2];
  const float dx = rd[ray*3+0], dy = rd[ray*3+1], dz = rd[ray*3+2];
  const float vx = vd[ray*3+0], vy = vd[ray*3+1], vz = vd[ray*3+2];
  const float dnorm = sqrtf(dx*dx + dy*dy + dz*dz);
  const float vex = (dx == 0.f) ? 1e-6f : dx;
  const float vey = (dy == 0.f) ? 1e-6f : dy;
  const float vez = (dz == 0.f) ? 1e-6f : dz;
  const float rax = ( 1.f-ox)/vex, rbx = (-1.f-ox)/vex;
  const float ray2= ( 1.f-oy)/vey, rby = (-1.f-oy)/vey;
  const float raz = ( 1.f-oz)/vez, rbz = (-1.f-oz)/vez;
  float tmn = fmaxf(fmaxf(fminf(rax,rbx), fminf(ray2,rby)), fminf(raz,rbz));
  float tmx = fminf(fminf(fmaxf(rax,rbx), fmaxf(ray2,rby)), fmaxf(raz,rbz));
  tmn = fminf(fmaxf(tmn, 0.2f), 6.0f);
  tmx = fminf(fmaxf(tmx, 0.2f), 6.0f);
  const bool maskray = (tmx <= tmn);
  const float stepw = 0.5f * (2.0f / 64.0f) / dnorm;

  float aR=0,aG=0,aB=0,aD=0,aA=0,aNx=0,aNy=0,aNz=0,aSx=0,aSy=0,aSz=0;
  float T = 1.0f;   // segment-local transmittance

  for (int it = 0; it < 4; ++it) {
    const int s = (wave * 4 + it) * 32 + lane;
    const float tt = tmn + stepw * (float)s;
    const float px = ox + dx*tt, py = oy + dy*tt, pz = oz + dz*tt;
    const bool inside = (!maskray) && (s < 447) &&
        (px >= -1.f) && (px <= 1.f) && (py >= -1.f) && (py <= 1.f) && (pz >= -1.f) && (pz <= 1.f);

    float ux = fminf(fmaxf((px + 1.f) * 63.5f, 0.f), 127.f);
    float uy = fminf(fmaxf((py + 1.f) * 63.5f, 0.f), 127.f);
    float uz = fminf(fmaxf((pz + 1.f) * 63.5f, 0.f), 127.f);
    int ix = (int)floorf(ux); ix = ix < 0 ? 0 : (ix > 126 ? 126 : ix);
    int iy = (int)floorf(uy); iy = iy < 0 ? 0 : (iy > 126 ? 126 : iy);
    int iz = (int)floorf(uz); iz = iz < 0 ? 0 : (iz > 126 ? 126 : iz);
    const float fx = ux - (float)ix, fy = uy - (float)iy, fz = uz - (float)iz;

    // prefetch next iteration's corner lines (global_prefetch_b8)
    {
      const float tt2 = tt + stepw * 32.f;
      const float qx = fminf(fmaxf((ox + dx*tt2 + 1.f) * 63.5f, 0.f), 126.f);
      const float qy = fminf(fmaxf((oy + dy*tt2 + 1.f) * 63.5f, 0.f), 126.f);
      const float qz = fminf(fmaxf((oz + dz*tt2 + 1.f) * 63.5f, 0.f), 126.f);
      const long fi2 = (((long)qx) * 128 + (long)qy) * 128 + (long)qz;
      __builtin_prefetch((const void*)(feat + fi2 * 16), 0, 0);
      __builtin_prefetch((const void*)(nrm + fi2 * 4), 0, 0);
    }

    float fv[16];
#pragma unroll
    for (int cc = 0; cc < 16; ++cc) fv[cc] = 0.f;
    float gx=0.f, gy=0.f, gz=0.f, sxn=0.f, syn=0.f, szn=0.f;
#pragma unroll
    for (int cr = 0; cr < 8; ++cr) {
      const int cdx = (cr >> 2) & 1, cdy = (cr >> 1) & 1, cdz = cr & 1;
      const long fi = ((long)(ix + cdx) * 128 + (iy + cdy)) * 128 + (iz + cdz);
      const v8h lo = *(const v8h*)(feat + fi * 16);
      const v8h hi = *(const v8h*)(feat + fi * 16 + 8);
      const v4h nv = *(const v4h*)(nrm + fi * 4);
      const float wx = cdx ? fx : 1.f - fx;
      const float wy = cdy ? fy : 1.f - fy;
      const float wz = cdz ? fz : 1.f - fz;
      const float w = wx * wy * wz;
      const float dens = (float)lo[0];
#pragma unroll
      for (int e = 0; e < 8; ++e) { fv[e] += w * (float)lo[e]; fv[8+e] += w * (float)hi[e]; }
      gx += (cdx ? 1.f : -1.f) * wy * wz * dens;
      gy += wx * (cdy ? 1.f : -1.f) * wz * dens;
      gz += wx * wy * (cdz ? 1.f : -1.f) * dens;
      sxn += w * (float)nv[0]; syn += w * (float)nv[1]; szn += w * (float)nv[2];
    }

    const float dsh = fv[0] - 9.2102404f;                       // + ACT_SHIFT
    const float sp = fmaxf(dsh, 0.f) + log1pf(__expf(-fabsf(dsh)));
    const float alpha = inside ? (1.f - __expf(-sp * 0.5f)) : 0.f;

    const float gl = sqrtf(gx*gx + gy*gy + gz*gz);
    const float gi = -1.f / fmaxf(gl, 1e-12f);
    const float ndx = inside ? gx*gi : 0.f;
    const float ndy = inside ? gy*gi : 0.f;
    const float ndz = inside ? gz*gi : 0.f;
    const float sl = sqrtf(sxn*sxn + syn*syn + szn*szn);
    const float si = -1.f / fmaxf(sl, 1e-12f);
    const float nsx = inside ? sxn*si : 0.f;
    const float nsy = inside ? syn*si : 0.f;
    const float nsz = inside ? szn*si : 0.f;

    const float dotv = -(vx*ndx + vy*ndy + vz*ndz);
    const float rx = 2.f*dotv*ndx + vx;
    const float ry = 2.f*dotv*ndy + vy;
    const float rz = 2.f*dotv*ndz + vz;

    // hfeat = [latent(15), refdirs(3), sin/cos x 6 freqs (36)] -> LDS row `lane`
    _Float16* hp = &hbuf[wave][lane][0];
#pragma unroll
    for (int cc = 0; cc < 15; ++cc) hp[cc] = (_Float16)fv[cc + 1];
    hp[15] = (_Float16)rx; hp[16] = (_Float16)ry; hp[17] = (_Float16)rz;
    {
      float s1x=__sinf(rx), c1x=__cosf(rx);
      float s1y=__sinf(ry), c1y=__cosf(ry);
      float s1z=__sinf(rz), c1z=__cosf(rz);
#pragma unroll
      for (int fq = 0; fq < 6; ++fq) {
        const int bse = 18 + fq * 6;
        hp[bse+0]=(_Float16)s1x; hp[bse+1]=(_Float16)s1y; hp[bse+2]=(_Float16)s1z;
        hp[bse+3]=(_Float16)c1x; hp[bse+4]=(_Float16)c1y; hp[bse+5]=(_Float16)c1z;
        float t2;
        t2 = 2.f*s1x*c1x; c1x = c1x*c1x - s1x*s1x; s1x = t2;
        t2 = 2.f*s1y*c1y; c1y = c1y*c1y - s1y*s1y; s1y = t2;
        t2 = 2.f*s1z*c1z; c1z = c1z*c1z - s1z*s1z; s1z = t2;
      }
    }
#pragma unroll
    for (int cc = 54; cc < 64; ++cc) hp[cc] = (_Float16)0.0f;
    __syncthreads();

    // MLP layers 0..2 (in-place LDS activation staging)
#pragma unroll
    for (int L = 0; L < 3; ++L) {
      v16h bf[2][2];
#pragma unroll
      for (int g = 0; g < 2; ++g)
#pragma unroll
        for (int ks = 0; ks < 2; ++ks)
          bf[g][ks] = *(const v16h*)&hbuf[wave][g*16 + n16][ks*32 + kh*16];
      __syncthreads();
#pragma unroll
      for (int g = 0; g < 2; ++g) {
        v8f cacc[4];
#pragma unroll
        for (int mt = 0; mt < 4; ++mt)
#pragma unroll
          for (int r = 0; r < 8; ++r) cacc[mt][r] = bias[L*64 + mt*16 + kh8 + r];
#pragma unroll
        for (int ks = 0; ks < 2; ++ks)
#pragma unroll
          for (int mt = 0; mt < 4; ++mt) {
            v16h a = *(const v16h*)&wf[L*8 + mt*2 + ks][lane][0];
            cacc[mt] = WMMA16(a, bf[g][ks], cacc[mt]);
          }
#pragma unroll
        for (int mt = 0; mt < 4; ++mt) {
          v8h hv;
#pragma unroll
          for (int r = 0; r < 8; ++r) hv[r] = (_Float16)fmaxf(cacc[mt][r], 0.f);
          *(v8h*)&hbuf[wave][g*16 + n16][mt*16 + kh8] = hv;
        }
      }
      __syncthreads();
    }

    // layer 3 -> sigmoid rgb
    float rgb0[3], rgb1[3];
#pragma unroll
    for (int g = 0; g < 2; ++g) {
      v8f c3a;
#pragma unroll
      for (int r = 0; r < 8; ++r) c3a[r] = bias[192 + kh8 + r];
#pragma unroll
      for (int ks = 0; ks < 2; ++ks) {
        v16h bfx = *(const v16h*)&hbuf[wave][g*16 + n16][ks*32 + kh*16];
        v16h a = *(const v16h*)&wf[24 + ks][lane][0];
        c3a = WMMA16(a, bfx, c3a);
      }
#pragma unroll
      for (int j = 0; j < 3; ++j) {
        const float val = 1.f / (1.f + __expf(-c3a[j]));
        if (g == 0) rgb0[j] = val; else rgb1[j] = val;
      }
    }

    // gather rgb of own sample (lane = sample index within group of 32)
    const float q0 = __shfl(rgb1[0], n16);
    const float q1 = __shfl(rgb1[1], n16);
    const float q2 = __shfl(rgb1[2], n16);
    const float rr = (lane < 16) ? rgb0[0] : q0;
    const float rg = (lane < 16) ? rgb0[1] : q1;
    const float rb = (lane < 16) ? rgb0[2] : q2;

    // transmittance scan (inclusive product) across 32 samples
    const float q = fmaxf(1.f - alpha, 1e-10f);
    float P = q;
#pragma unroll
    for (int off = 1; off < 32; off <<= 1) {
      const float pv = __shfl_up(P, off);
      if (lane >= off) P *= pv;
    }
    float Pex = __shfl_up(P, 1);
    if (lane == 0) Pex = 1.f;
    const float w = alpha * T * Pex;
    aR += w*rr; aG += w*rg; aB += w*rb;
    aD += w * tt * dnorm;
    aA += w;
    aNx += w*ndx; aNy += w*ndy; aNz += w*ndz;
    aSx += w*nsx; aSy += w*nsy; aSz += w*nsz;
    T *= __shfl(P, 31);
    __syncthreads();
  }

  // per-segment lane reduction
#pragma unroll
  for (int off = 16; off >= 1; off >>= 1) {
    aR += __shfl_xor(aR, off); aG += __shfl_xor(aG, off); aB += __shfl_xor(aB, off);
    aD += __shfl_xor(aD, off); aA += __shfl_xor(aA, off);
    aNx += __shfl_xor(aNx, off); aNy += __shfl_xor(aNy, off); aNz += __shfl_xor(aNz, off);
    aSx += __shfl_xor(aSx, off); aSy += __shfl_xor(aSy, off); aSz += __shfl_xor(aSz, off);
  }
  if (lane == 0) {
    float* cp = comb[wave];
    cp[0]=aR; cp[1]=aG; cp[2]=aB; cp[3]=aD; cp[4]=aA;
    cp[5]=aNx; cp[6]=aNy; cp[7]=aNz; cp[8]=aSx; cp[9]=aSy; cp[10]=aSz; cp[11]=T;
  }
  __syncthreads();

  // segmented composition: out = sum_k G_k * S_k, G_k = prod_{j<k} T_j
  if (tid == 0) {
    float sm[11];
#pragma unroll
    for (int j = 0; j < 11; ++j) sm[j] = 0.f;
    float G = 1.f;
#pragma unroll
    for (int sg = 0; sg < 4; ++sg) {
#pragma unroll
      for (int j = 0; j < 11; ++j) sm[j] += G * comb[sg][j];
      G *= comb[sg][11];
    }
    const float dep = sm[3] + G * 6.0f;
    out[ray*12+0] = sm[0] + G;
    out[ray*12+1] = sm[1] + G;
    out[ray*12+2] = sm[2] + G;
    out[ray*12+3] = dep;
    out[ray*12+4] = 1.f / dep;
    out[ray*12+5] = sm[4];
    out[ray*12+6] = sm[5]; out[ray*12+7] = sm[6]; out[ray*12+8] = sm[7];
    out[ray*12+9] = sm[8]; out[ray*12+10] = sm[9]; out[ray*12+11] = sm[10];
  }
}

extern "C" void kernel_launch(void* const* d_in, const int* in_sizes, int n_in,
                              void* d_out, int out_size, void* d_ws, size_t ws_size,
                              hipStream_t stream) {
  const float* ro   = (const float*)d_in[0];
  const float* rdp  = (const float*)d_in[1];
  const float* vdp  = (const float*)d_in[2];
  const float* grid = (const float*)d_in[3];
  const float* c1w = (const float*)d_in[4];  const float* c1b = (const float*)d_in[5];
  const float* g1  = (const float*)d_in[6];  const float* bb1 = (const float*)d_in[7];
  const float* a1  = (const float*)d_in[8];
  const float* c2w = (const float*)d_in[9];  const float* c2b = (const float*)d_in[10];
  const float* g2  = (const float*)d_in[11]; const float* bb2 = (const float*)d_in[12];
  const float* a2  = (const float*)d_in[13];
  const float* c3w = (const float*)d_in[14]; const float* c3b = (const float*)d_in[15];
  const float* w0 = (const float*)d_in[16];  const float* b0 = (const float*)d_in[17];
  const float* w1 = (const float*)d_in[18];  const float* b1 = (const float*)d_in[19];
  const float* w2 = (const float*)d_in[20];  const float* b2 = (const float*)d_in[21];
  const float* w3 = (const float*)d_in[22];  const float* b3 = (const float*)d_in[23];

  _Float16* feat = (_Float16*)d_ws;                                        // 128^3*16 f16 = 64MB
  _Float16* nrm  = (_Float16*)((char*)d_ws + (size_t)128*128*128*16*2);    // 128^3*4  f16 = 16MB

  k_gridreg<<<dim3(2048), dim3(256), 0, stream>>>(grid, c1w, c1b, g1, bb1, a1,
                                                  c2w, c2b, g2, bb2, a2, c3w, c3b, feat);
  k_sobel<<<dim3(4096), dim3(512), 0, stream>>>(feat, nrm);
  k_render<<<dim3(1024), dim3(128), 0, stream>>>(ro, rdp, vdp, feat, nrm,
                                                 w0, b0, w1, b1, w2, b2, w3, b3,
                                                 (float*)d_out);
}
